// Head_1649267441819
// MI455X (gfx1250) — matrix operand order repaired and checked by reading, not compiled, not run on hardware
//
#include <hip/hip_runtime.h>

// ---------------------------------------------------------------------------
// Single-head causal attention (B=8, T=2048, C=1024, H=64) for gfx1250.
// f16 WMMA (f32 accumulate) flash-attention; Sᵀ trick keeps P in registers.
// Projection loads each x tile once and feeds 12 WMMAs (3 mats x 4 cols).
// ---------------------------------------------------------------------------

typedef __attribute__((ext_vector_type(16))) _Float16 v16h;
typedef __attribute__((ext_vector_type(8)))  _Float16 v8h;
typedef __attribute__((ext_vector_type(8)))  float    v8f;

#define N_EMBD 1024
#define HEADSZ 64
#define SEQ    2048
#define NBATCH 8

static __device__ __forceinline__ v8f wmma_f16(v16h a, v16h b, v8f c) {
    return __builtin_amdgcn_wmma_f32_16x16x32_f16(false, a, false, b,
                                                  (short)0, c, false, false);
}

// ---------------------------------------------------------------------------
// Kernel 1: W (C x H, f32 row-major) -> Wt (H x C, f16)  [3 matrices: K,Q,V]
// ---------------------------------------------------------------------------
__global__ void prep_weights(const float* __restrict__ Wk,
                             const float* __restrict__ Wq,
                             const float* __restrict__ Wv,
                             _Float16* __restrict__ Wt) {
    int idx = blockIdx.x * blockDim.x + threadIdx.x;
    if (idx >= 3 * HEADSZ * N_EMBD) return;
    int mat = idx >> 16;            // 65536 elems per matrix
    int r   = idx & 65535;
    int h   = r >> 10;
    int c   = r & 1023;
    const float* W = (mat == 0) ? Wk : (mat == 1) ? Wq : Wv;
    Wt[idx] = (_Float16)W[c * HEADSZ + h];
}

// ---------------------------------------------------------------------------
// Kernel 2: projections via WMMA with full A-tile reuse.
// One wave owns a 16-row tile of x and produces ALL 12 output tiles
// (K,Q,V x 4 head-column tiles): 12 accumulators, x read exactly once.
//   Q,K stored row-major f16 [B][T][H] ; V stored transposed f16 [B][H][T].
// ---------------------------------------------------------------------------
__global__ void proj_kernel(const float* __restrict__ x,
                            const _Float16* __restrict__ Wt,
                            _Float16* __restrict__ Kb,
                            _Float16* __restrict__ Qb,
                            _Float16* __restrict__ Vt) {
    const int lane = threadIdx.x & 31;
    const int gw   = blockIdx.x * (blockDim.x >> 5) + (threadIdx.x >> 5);
    const int b    = gw >> 7;
    const int t0   = (gw & 127) << 4;
    const int lm   = lane & 15;
    const int hi   = lane >> 4;            // wave half (A/B K-subrange select)

    const float* xrow = x + (size_t)(b * SEQ + t0 + lm) * N_EMBD;

    v8f acc[12];
#pragma unroll
    for (int i = 0; i < 12; ++i) acc[i] = (v8f){};

    for (int k = 0; k < N_EMBD / 32; ++k) {
        const int c0 = k * 32;
        // A tile (16x32 of x), loaded once, converted f32->f16.
        v8f x0 = *(const v8f*)(xrow + c0 + hi * 16);
        v8f x1 = *(const v8f*)(xrow + c0 + hi * 16 + 8);
        v16h a;
#pragma unroll
        for (int i = 0; i < 8; ++i) { a[i] = (_Float16)x0[i]; a[8 + i] = (_Float16)x1[i]; }
        // 12 B tiles from the (L2-resident) transposed weights; 12 WMMAs.
#pragma unroll
        for (int mat = 0; mat < 3; ++mat) {
#pragma unroll
            for (int col = 0; col < 4; ++col) {
                const _Float16* wp = Wt + (size_t)mat * (HEADSZ * N_EMBD)
                                        + (size_t)(col * 16 + lm) * N_EMBD
                                        + c0 + hi * 16;
                v16h bm = *(const v16h*)wp;
                acc[mat * 4 + col] = wmma_f16(a, bm, acc[mat * 4 + col]);
            }
        }
    }

#pragma unroll
    for (int col = 0; col < 4; ++col) {
        const int h = col * 16 + lm;
        // K (mat 0) and Q (mat 1): row-major f16 stores.
#pragma unroll
        for (int j = 0; j < 8; ++j) {
            const int t = t0 + hi * 8 + j;
            Kb[(size_t)(b * SEQ + t) * HEADSZ + h] = (_Float16)acc[0 + col][j];
            Qb[(size_t)(b * SEQ + t) * HEADSZ + h] = (_Float16)acc[4 + col][j];
        }
        // V (mat 2): transposed layout -> lane's 8 rows contiguous in t.
        v8h pk;
#pragma unroll
        for (int j = 0; j < 8; ++j) pk[j] = (_Float16)acc[8 + col][j];
        *(v8h*)(Vt + (size_t)(b * HEADSZ + h) * SEQ + t0 + hi * 8) = pk;
    }
}

// ---------------------------------------------------------------------------
// Kernel 3: flash attention. One wave owns 16 query rows, loops key blocks
// of 32. Computes Sᵀ = K·Qᵀ so softmax output P lands directly in the
// 16-bit A-matrix layout for the P·V WMMA (no cross-lane transpose).
// Softmax runs in the exp2 domain (v_exp_f32 is natively base-2).
// ---------------------------------------------------------------------------
__global__ void attn_kernel(const _Float16* __restrict__ Qb,
                            const _Float16* __restrict__ Kb,
                            const _Float16* __restrict__ Vt,
                            float* __restrict__ out) {
    const int lane = threadIdx.x & 31;
    const int gw   = blockIdx.x * (blockDim.x >> 5) + (threadIdx.x >> 5);
    const int b    = gw >> 7;
    const int q0   = (gw & 127) << 4;
    const int lm   = lane & 15;
    const int hi   = lane >> 4;

    const float scale2 = 0.125f * 1.44269504089f;   // (1/sqrt(H)) * log2(e)

    // Qᵀ as B-matrix (K = head dim): 2 tiles covering h=0..31, 32..63.
    const _Float16* qrow = Qb + (size_t)(b * SEQ + q0 + lm) * HEADSZ;
    const v16h bq0 = *(const v16h*)(qrow + 0  + hi * 16);
    const v16h bq1 = *(const v16h*)(qrow + 32 + hi * 16);

    v8f o0 = {}, o1 = {}, o2 = {}, o3 = {};
    float m = -1e30f, l = 0.0f;

    const int qidx = q0 + lm;             // this lane's query row
    const int nblk = (q0 + 47) >> 5;      // key blocks with s0 <= q0+15

    for (int kb = 0; kb < nblk; ++kb) {
        const int s0 = kb << 5;
        // ---- Sᵀ = K · Qᵀ : two 16(key)x16(query) sub-tiles ----
        v8f sa = {}, sb = {};
        const _Float16* krow0 = Kb + (size_t)(b * SEQ + s0 + lm) * HEADSZ;
        const _Float16* krow1 = krow0 + (size_t)16 * HEADSZ;
#pragma unroll
        for (int c = 0; c < 2; ++c) {
            const int koff = c * 32 + hi * 16;
            const v16h bqc = (c == 0) ? bq0 : bq1;
            v8h lo = *(const v8h*)(krow0 + koff);
            v8h hh = *(const v8h*)(krow0 + koff + 8);
            v16h aK;
#pragma unroll
            for (int i = 0; i < 8; ++i) { aK[i] = lo[i]; aK[8 + i] = hh[i]; }
            sa = wmma_f16(aK, bqc, sa);
            lo = *(const v8h*)(krow1 + koff);
            hh = *(const v8h*)(krow1 + koff + 8);
#pragma unroll
            for (int i = 0; i < 8; ++i) { aK[i] = lo[i]; aK[8 + i] = hh[i]; }
            sb = wmma_f16(aK, bqc, sb);
        }

        // ---- scale (log2 domain) + causal mask + online softmax ----
        // Lane holds query row qidx; keys: sa -> s0+hi*8+i, sb -> +16.
        const int kbase = s0 + hi * 8;
        float pa[8], pb[8];
        float vmax = -1e30f;
#pragma unroll
        for (int i = 0; i < 8; ++i) {
            float va = (kbase + i      <= qidx) ? sa[i] * scale2 : -1e30f;
            float vb = (kbase + 16 + i <= qidx) ? sb[i] * scale2 : -1e30f;
            pa[i] = va; pb[i] = vb;
            vmax = fmaxf(vmax, fmaxf(va, vb));
        }
        vmax = fmaxf(vmax, __shfl_xor(vmax, 16, 32));   // merge lane-pair halves
        const float m_new = fmaxf(m, vmax);
        const float alpha = __builtin_exp2f(m - m_new);

        float rsum = 0.0f;
        v16h P;                                          // A-layout P tile
#pragma unroll
        for (int i = 0; i < 8; ++i) {
            float ea = __builtin_exp2f(pa[i] - m_new);
            float eb = __builtin_exp2f(pb[i] - m_new);
            rsum += ea + eb;
            P[i] = (_Float16)ea; P[8 + i] = (_Float16)eb;
        }
        rsum += __shfl_xor(rsum, 16, 32);
        l = l * alpha + rsum;
        m = m_new;

        // ---- rescale O (row j stats live at lane j / j+8) ----
#pragma unroll
        for (int j = 0; j < 8; ++j) {
            float aj = __shfl(alpha, hi ? (j + 8) : j, 32);
            o0[j] *= aj; o1[j] *= aj; o2[j] *= aj; o3[j] *= aj;
        }

        // ---- O += P · V ; V columns contiguous in transposed layout ----
        const _Float16* vb = Vt + (size_t)b * HEADSZ * SEQ + s0 + hi * 8;
        v16h bv; v8h v0, v1;
        v0 = *(const v8h*)(vb + (size_t)(lm +  0) * SEQ); v1 = *(const v8h*)(vb + (size_t)(lm +  0) * SEQ + 16);
        bv = __builtin_shufflevector(v0, v1, 0,1,2,3,4,5,6,7,8,9,10,11,12,13,14,15); o0 = wmma_f16(P, bv, o0);
        v0 = *(const v8h*)(vb + (size_t)(lm + 16) * SEQ); v1 = *(const v8h*)(vb + (size_t)(lm + 16) * SEQ + 16);
        bv = __builtin_shufflevector(v0, v1, 0,1,2,3,4,5,6,7,8,9,10,11,12,13,14,15); o1 = wmma_f16(P, bv, o1);
        v0 = *(const v8h*)(vb + (size_t)(lm + 32) * SEQ); v1 = *(const v8h*)(vb + (size_t)(lm + 32) * SEQ + 16);
        bv = __builtin_shufflevector(v0, v1, 0,1,2,3,4,5,6,7,8,9,10,11,12,13,14,15); o2 = wmma_f16(P, bv, o2);
        v0 = *(const v8h*)(vb + (size_t)(lm + 48) * SEQ); v1 = *(const v8h*)(vb + (size_t)(lm + 48) * SEQ + 16);
        bv = __builtin_shufflevector(v0, v1, 0,1,2,3,4,5,6,7,8,9,10,11,12,13,14,15); o3 = wmma_f16(P, bv, o3);
    }

    // ---- finalize: divide by row sum, store f32 [B][T][H] ----
#pragma unroll
    for (int j = 0; j < 8; ++j) {
        float lj  = __shfl(l, hi ? (j + 8) : j, 32);
        float inv = 1.0f / lj;
        const int t = q0 + hi * 8 + j;
        float* orow = out + (size_t)(b * SEQ + t) * HEADSZ + lm;
        orow[0]  = o0[j] * inv;
        orow[16] = o1[j] * inv;
        orow[32] = o2[j] * inv;
        orow[48] = o3[j] * inv;
    }
}

// ---------------------------------------------------------------------------
extern "C" void kernel_launch(void* const* d_in, const int* in_sizes, int n_in,
                              void* d_out, int out_size, void* d_ws, size_t ws_size,
                              hipStream_t stream) {
    const float* x  = (const float*)d_in[0];
    const float* Wk = (const float*)d_in[1];
    const float* Wq = (const float*)d_in[2];
    const float* Wv = (const float*)d_in[3];
    float* out = (float*)d_out;

    // workspace carve-up (all f16)
    char* ws = (char*)d_ws;
    _Float16* Wt  = (_Float16*)(ws);                               // 3*64K halves
    _Float16* Kb  = (_Float16*)(ws + 3 * 65536 * 2);               // B*T*H
    _Float16* Qb  = (_Float16*)(ws + 3 * 65536 * 2 + 2097152);
    _Float16* Vt  = (_Float16*)(ws + 3 * 65536 * 2 + 2 * 2097152);

    // 1) weight convert/transpose
    prep_weights<<<(3 * HEADSZ * N_EMBD + 255) / 256, 256, 0, stream>>>(Wk, Wq, Wv, Wt);

    // 2) projections: one wave per 16-row x tile -> B*T/16 = 1024 waves
    proj_kernel<<<1024 / 8, 256, 0, stream>>>(x, Wt, Kb, Qb, Vt);

    // 3) flash attention: B * T/16 = 1024 waves
    attn_kernel<<<1024 / 8, 256, 0, stream>>>(Qb, Kb, Vt, out);
}
